// CrossTransformerEncoder_19138374271741
// MI455X (gfx1250) — matrix operand order repaired and checked by reading, not compiled, not run on hardware
//
#include <hip/hip_runtime.h>
#include <hip/hip_bf16.h>

// ---------------------------------------------------------------------------
// CDNA5 (gfx1250) bf16 WMMA implementation of CrossTransformerEncoder.
// wave32; v_wmma_f32_16x16x32_bf16; 128x128x32 block tiles, 8 waves/block,
// each wave computing 32x64 (8 accumulators) from double-buffered LDS.
// ---------------------------------------------------------------------------

typedef __bf16 bf16_t;
typedef __bf16 v16bf __attribute__((ext_vector_type(16)));
typedef __bf16 v8bf  __attribute__((ext_vector_type(8)));
typedef float  v8f   __attribute__((ext_vector_type(8)));

#define THREADS 256
#define BM 128
#define BN 128
#define BK 32
#define LDP 40  // padded LDS row stride (halves): 80B/row -> conflict-free frags

enum { ACT_NONE = 0, ACT_ELU1 = 1, ACT_RELU = 2 };

struct StageRegs { uint4 d[2]; };

// Load one 128x32 bf16 tile's worth (16B x 2 per thread) from global into regs.
// T=false: element (r,c) at src[(base+r)*ld + k0 + c]  (row-major)
// T=true : element (r,c) at src[(k0+c)*ld + base + r]  (transposed view)
template <bool T>
__device__ __forceinline__ StageRegs stage_load(const bf16_t* __restrict__ src,
                                                long ld, int base, int k0, int tid) {
  StageRegs r;
#pragma unroll
  for (int jj = 0; jj < 2; ++jj) {
    int id = tid + jj * 256;
    if (!T) {
      int row = id >> 2, cg = (id & 3) << 3;
      r.d[jj] = *reinterpret_cast<const uint4*>(src + (long)(base + row) * ld + k0 + cg);
    } else {
      int c = id >> 4, rg = (id & 15) << 3;
      r.d[jj] = *reinterpret_cast<const uint4*>(src + (long)(k0 + c) * ld + base + rg);
    }
  }
  return r;
}

template <bool T>
__device__ __forceinline__ void stage_store(bf16_t* lds, StageRegs r, int tid) {
#pragma unroll
  for (int jj = 0; jj < 2; ++jj) {
    int id = tid + jj * 256;
    if (!T) {
      int row = id >> 2, cg = (id & 3) << 3;
      *reinterpret_cast<uint4*>(&lds[row * LDP + cg]) = r.d[jj];
    } else {
      int c = id >> 4, rg = (id & 15) << 3;
      bf16_t tmp[8];
      *reinterpret_cast<uint4*>(tmp) = r.d[jj];
#pragma unroll
      for (int j2 = 0; j2 < 8; ++j2) lds[(rg + j2) * LDP + c] = tmp[j2];
    }
  }
}

// ---------------------------------------------------------------------------
// C[m,n] = act(sum_k A[m,k]*Bw[n,k]) * cscale * rowscale[m]
//   AT/BT: operand stored transposed in memory; ST: store C transposed;
//   OF32: f32 output (else bf16).
// Grid: (N/BN, M/BM, batch); block: 256 threads (8 wave32).
// ---------------------------------------------------------------------------
template <int ACT, bool AT, bool BT, bool ST, bool OF32>
__global__ __launch_bounds__(THREADS) void gemm_wmma(
    const bf16_t* __restrict__ A, long lda, long strideA,
    const bf16_t* __restrict__ B, long ldb, long strideB,
    void* __restrict__ Cout, long ldc, long strideC,
    const float* __restrict__ rowscale, long rsStride, float cscale, int K) {
  __shared__ bf16_t lA[2][BM * LDP];
  __shared__ bf16_t lB[2][BN * LDP];

  const int tid  = threadIdx.x;
  const int lane = tid & 31;
  const int wave = tid >> 5;
  const int wm2  = wave & 3;   // 32-row band: rows wm2*32 .. +31
  const int wg   = wave >> 2;  // 64-col band: cols wg*64 .. +63
  const int batch = blockIdx.z;
  const int mb = blockIdx.y * BM;
  const int nb = blockIdx.x * BN;

  A += (long)batch * strideA;
  B += (long)batch * strideB;

  v8f acc[2][4];
#pragma unroll
  for (int i = 0; i < 2; ++i)
#pragma unroll
    for (int j = 0; j < 4; ++j) acc[i][j] = {0.f, 0.f, 0.f, 0.f, 0.f, 0.f, 0.f, 0.f};

  const int nk = K / BK;

  StageRegs ra = stage_load<AT>(A, lda, mb, 0, tid);
  StageRegs rb = stage_load<BT>(B, ldb, nb, 0, tid);
  stage_store<AT>(lA[0], ra, tid);
  stage_store<BT>(lB[0], rb, tid);
  __syncthreads();

  for (int kt = 0; kt < nk; ++kt) {
    const int cur = kt & 1;

    // issue next tile's global loads early (latency hidden under WMMA)
    if (kt + 1 < nk) {
      ra = stage_load<AT>(A, lda, mb, (kt + 1) * BK, tid);
      rb = stage_load<BT>(B, ldb, nb, (kt + 1) * BK, tid);
    }
    // warm GL2 one further step ahead
    if (kt + 2 < nk) {
      int k2 = (kt + 2) * BK;
      if (!AT) __builtin_prefetch(A + (long)(mb + (tid >> 2)) * lda + k2 + ((tid & 3) << 3), 0, 1);
      else     __builtin_prefetch(A + (long)(k2 + (tid >> 4)) * lda + mb + ((tid & 15) << 3), 0, 1);
      if (!BT) __builtin_prefetch(B + (long)(nb + (tid >> 2)) * ldb + k2 + ((tid & 3) << 3), 0, 1);
      else     __builtin_prefetch(B + (long)(k2 + (tid >> 4)) * ldb + nb + ((tid & 15) << 3), 0, 1);
    }

    // ---- fragments per ISA layout ----
    union Frag { v16bf v; v8bf h[2]; };
    Frag af[2], bfr[4];
#pragma unroll
    for (int i = 0; i < 2; ++i) {
      // A 16x32: lane<16 -> M=lane, K chunks {c0..c0+7, c0+16..c0+23}, c0=(lane/16)*8
      int r  = wm2 * 32 + i * 16 + (lane & 15);
      int c0 = (lane >> 4) * 8;
      af[i].h[0] = *reinterpret_cast<const v8bf*>(&lA[cur][r * LDP + c0]);
      af[i].h[1] = *reinterpret_cast<const v8bf*>(&lA[cur][r * LDP + c0 + 16]);
    }
#pragma unroll
    for (int j = 0; j < 4; ++j) {
      // B 32x16: lane holds column N=lane%16, K chunks {kk..kk+7, kk+16..kk+23}, kk=(lane/16)*8
      int r  = wg * 64 + j * 16 + (lane & 15);
      int kk = (lane >> 4) * 8;
      bfr[j].h[0] = *reinterpret_cast<const v8bf*>(&lB[cur][r * LDP + kk]);
      bfr[j].h[1] = *reinterpret_cast<const v8bf*>(&lB[cur][r * LDP + kk + 16]);
    }
#pragma unroll
    for (int i = 0; i < 2; ++i)
#pragma unroll
      for (int j = 0; j < 4; ++j)
        acc[i][j] = __builtin_amdgcn_wmma_f32_16x16x32_bf16(
            false, af[i].v, false, bfr[j].v, (short)0, acc[i][j], false, false);

    // stash next tile into the other LDS buffer after the math is issued
    if (kt + 1 < nk) {
      stage_store<AT>(lA[1 - cur], ra, tid);
      stage_store<BT>(lB[1 - cur], rb, tid);
    }
    __syncthreads();
  }

  // ---- epilogue: C/D layout N=lane%16, M=(lane/16)*8 + r ----
  if (rowscale) rowscale += (long)batch * rsStride;
  const int ncol  = lane & 15;
  const int moff  = (lane >> 4) * 8;

  float rsv[2][8];
#pragma unroll
  for (int i = 0; i < 2; ++i)
#pragma unroll
    for (int r = 0; r < 8; ++r)
      rsv[i][r] = rowscale ? rowscale[mb + wm2 * 32 + i * 16 + moff + r] : 1.0f;

#pragma unroll
  for (int i = 0; i < 2; ++i) {
#pragma unroll
    for (int j = 0; j < 4; ++j) {
      int ng = nb + wg * 64 + j * 16 + ncol;
#pragma unroll
      for (int r = 0; r < 8; ++r) {
        int mg = mb + wm2 * 32 + i * 16 + moff + r;
        float v = acc[i][j][r];
        if (ACT == ACT_ELU1) v = (v > 0.f) ? (v + 1.f) : __expf(v);  // elu(v)+1
        else if (ACT == ACT_RELU) v = (v > 0.f) ? v : 0.f;
        v *= cscale * rsv[i][r];
        long idx = ST ? ((long)ng * ldc + mg) : ((long)mg * ldc + ng);
        if (OF32)
          reinterpret_cast<float*>(Cout)[(long)batch * strideC + idx] = v;
        else
          reinterpret_cast<bf16_t*>(Cout)[(long)batch * strideC + idx] = (bf16_t)v;
      }
    }
  }
}

// ---------------------------------------------------------------------------
// Elementwise / reduction helpers
// ---------------------------------------------------------------------------
__global__ void cvt_bf16(const float* __restrict__ s, bf16_t* __restrict__ d, long n) {
  long i = (long)blockIdx.x * blockDim.x + threadIdx.x;
  if (i < n) d[i] = (bf16_t)s[i];
}

__global__ void cvt_bf16_strided(const float* __restrict__ s, bf16_t* __restrict__ d,
                                 int dstLd, int dstOff, long n) {
  long i = (long)blockIdx.x * blockDim.x + threadIdx.x;
  if (i < n) {
    long row = i >> 9;  // /512
    int  c   = (int)(i & 511);
    d[row * dstLd + dstOff + c] = (bf16_t)s[i];
  }
}

// Ksum[n][d] = sum_s K[n][s][d]   (K: bf16 [64][1024][512])
__global__ void ksum_kernel(const bf16_t* __restrict__ K, float* __restrict__ Ksum) {
  int i = blockIdx.x * blockDim.x + threadIdx.x;  // 64*512 threads
  int n = i >> 9, d = i & 511;
  const bf16_t* p = K + (long)n * 1024 * 512 + d;
  float s = 0.f;
  for (int t = 0; t < 1024; ++t) s += (float)p[(long)t * 512];
  Ksum[i] = s;
}

// Z[m] = 1 / (dot(Q[m,:], Ksum[m/1024,:]) + eps); one wave32 per row
__global__ __launch_bounds__(256) void z_kernel(const bf16_t* __restrict__ Q,
                                                const float* __restrict__ Ksum,
                                                float* __restrict__ Z) {
  int lane = threadIdx.x & 31;
  int w    = threadIdx.x >> 5;
  long row = (long)blockIdx.x * 8 + w;  // 65536 rows
  int  n   = (int)(row >> 10);
  const bf16_t* q  = Q + row * 512;
  const float*  ks = Ksum + (long)n * 512;
  float s = 0.f;
  for (int d = lane; d < 512; d += 32) s += (float)q[d] * ks[d];
  for (int off = 16; off > 0; off >>= 1) s += __shfl_down(s, off, 32);
  if (lane == 0) Z[row] = 1.0f / (s + 1e-6f);
}

// LayerNorm over 512 cols; one 256-thread block per row -> bf16 strided out
__global__ __launch_bounds__(256) void ln_to_bf16(const float* __restrict__ in,
                                                  const float* __restrict__ g,
                                                  const float* __restrict__ b,
                                                  bf16_t* __restrict__ out,
                                                  int outLd, int outOff) {
  __shared__ float red[256];
  long row = blockIdx.x;
  const float* p = in + row * 512;
  int t = threadIdx.x;
  float v0 = p[t], v1 = p[t + 256];
  red[t] = v0 + v1;
  __syncthreads();
  for (int o = 128; o > 0; o >>= 1) {
    if (t < o) red[t] += red[t + o];
    __syncthreads();
  }
  float mu = red[0] * (1.f / 512.f);
  __syncthreads();
  float d0 = v0 - mu, d1 = v1 - mu;
  red[t] = d0 * d0 + d1 * d1;
  __syncthreads();
  for (int o = 128; o > 0; o >>= 1) {
    if (t < o) red[t] += red[t + o];
    __syncthreads();
  }
  float rs = rsqrtf(red[0] * (1.f / 512.f) + 1e-5f);
  out[row * outLd + outOff + t]       = (bf16_t)(d0 * rs * g[t] + b[t]);
  out[row * outLd + outOff + t + 256] = (bf16_t)(d1 * rs * g[t + 256] + b[t + 256]);
}

// Final LN + residual: out = x + (norm(h)*g + b), f32 out
__global__ __launch_bounds__(256) void ln_residual(const float* __restrict__ in,
                                                   const float* __restrict__ x,
                                                   const float* __restrict__ g,
                                                   const float* __restrict__ b,
                                                   float* __restrict__ out) {
  __shared__ float red[256];
  long row = blockIdx.x;
  const float* p = in + row * 512;
  int t = threadIdx.x;
  float v0 = p[t], v1 = p[t + 256];
  red[t] = v0 + v1;
  __syncthreads();
  for (int o = 128; o > 0; o >>= 1) {
    if (t < o) red[t] += red[t + o];
    __syncthreads();
  }
  float mu = red[0] * (1.f / 512.f);
  __syncthreads();
  float d0 = v0 - mu, d1 = v1 - mu;
  red[t] = d0 * d0 + d1 * d1;
  __syncthreads();
  for (int o = 128; o > 0; o >>= 1) {
    if (t < o) red[t] += red[t + o];
    __syncthreads();
  }
  float rs = rsqrtf(red[0] * (1.f / 512.f) + 1e-5f);
  out[row * 512 + t]       = x[row * 512 + t]       + d0 * rs * g[t] + b[t];
  out[row * 512 + t + 256] = x[row * 512 + t + 256] + d1 * rs * g[t + 256] + b[t + 256];
}

// ---------------------------------------------------------------------------
// Host driver
// ---------------------------------------------------------------------------
extern "C" void kernel_launch(void* const* d_in, const int* in_sizes, int n_in,
                              void* d_out, int out_size, void* d_ws, size_t ws_size,
                              hipStream_t stream) {
  (void)in_sizes; (void)n_in; (void)out_size; (void)ws_size;
  const float* x    = (const float*)d_in[0];
  const float* src  = (const float*)d_in[1];
  const float* Wq   = (const float*)d_in[2];
  const float* Wk   = (const float*)d_in[3];
  const float* Wv   = (const float*)d_in[4];
  const float* Wm   = (const float*)d_in[5];
  const float* W1   = (const float*)d_in[6];
  const float* W2   = (const float*)d_in[7];
  const float* g1   = (const float*)d_in[8];
  const float* b1   = (const float*)d_in[9];
  const float* g2   = (const float*)d_in[10];
  const float* b2   = (const float*)d_in[11];
  const float* xm   = (const float*)d_in[12];
  const float* sm   = (const float*)d_in[13];
  float* out = (float*)d_out;

  const long NB = 64, L = 1024;
  const long R = NB * L;  // 65536 token rows

  size_t off = 0;
  auto carve = [&](size_t bytes) {
    size_t o = off;
    off += (bytes + 255) & ~(size_t)255;
    return (char*)d_ws + o;
  };
  bf16_t* hcat = (bf16_t*)carve(R * 1024 * 2);  // [65536][1024]: x | msgn
  bf16_t* srcb = (bf16_t*)carve(R * 512 * 2);
  bf16_t* Qb   = (bf16_t*)carve(R * 512 * 2);
  bf16_t* Kb   = (bf16_t*)carve(R * 512 * 2);
  bf16_t* Vb   = (bf16_t*)carve(R * 512 * 2);
  bf16_t* KVT  = (bf16_t*)carve(NB * 512 * 512 * 2);
  float*  Ksum = (float*)carve(NB * 512 * 4);
  float*  Zs   = (float*)carve(R * 4);
  bf16_t* msgt = (bf16_t*)carve(R * 512 * 2);
  float*  fbuf = (float*)carve(R * 512 * 4);
  bf16_t* h1   = (bf16_t*)carve(R * 1024 * 2);
  bf16_t* Wqb  = (bf16_t*)carve(512 * 512 * 2);
  bf16_t* Wkb  = (bf16_t*)carve(512 * 512 * 2);
  bf16_t* Wvb  = (bf16_t*)carve(512 * 512 * 2);
  bf16_t* Wmb  = (bf16_t*)carve(512 * 512 * 2);
  bf16_t* W1b  = (bf16_t*)carve(1024 * 1024 * 2);
  bf16_t* W2b  = (bf16_t*)carve(512 * 1024 * 2);

  dim3 blk(THREADS);
  auto cgrid = [](long n) { return dim3((unsigned)((n + 255) / 256)); };

  cvt_bf16<<<cgrid(512 * 512), blk, 0, stream>>>(Wq, Wqb, 512 * 512);
  cvt_bf16<<<cgrid(512 * 512), blk, 0, stream>>>(Wk, Wkb, 512 * 512);
  cvt_bf16<<<cgrid(512 * 512), blk, 0, stream>>>(Wv, Wvb, 512 * 512);
  cvt_bf16<<<cgrid(512 * 512), blk, 0, stream>>>(Wm, Wmb, 512 * 512);
  cvt_bf16<<<cgrid(1024 * 1024), blk, 0, stream>>>(W1, W1b, 1024 * 1024);
  cvt_bf16<<<cgrid(512 * 1024), blk, 0, stream>>>(W2, W2b, 512 * 1024);
  cvt_bf16<<<cgrid(R * 512), blk, 0, stream>>>(src, srcb, R * 512);
  cvt_bf16_strided<<<cgrid(R * 512), blk, 0, stream>>>(x, hcat, 1024, 0, R * 512);

  // Q = (elu(x@Wq^T)+1) * x_mask     [65536 x 512]
  gemm_wmma<ACT_ELU1, false, false, false, false>
      <<<dim3(512 / BN, R / BM, 1), blk, 0, stream>>>(
          hcat, 1024, 0, Wqb, 512, 0, Qb, 512, 0, xm, 0, 1.0f, 512);
  // K = (elu(src@Wk^T)+1) * src_mask
  gemm_wmma<ACT_ELU1, false, false, false, false>
      <<<dim3(512 / BN, R / BM, 1), blk, 0, stream>>>(
          srcb, 512, 0, Wkb, 512, 0, Kb, 512, 0, sm, 0, 1.0f, 512);
  // V = (src@Wv^T) * src_mask / 1024
  gemm_wmma<ACT_NONE, false, false, false, false>
      <<<dim3(512 / BN, R / BM, 1), blk, 0, stream>>>(
          srcb, 512, 0, Wvb, 512, 0, Vb, 512, 0, sm, 0, 1.0f / 1024.f, 512);

  ksum_kernel<<<cgrid(NB * 512), blk, 0, stream>>>(Kb, Ksum);
  z_kernel<<<dim3((unsigned)(R / 8)), blk, 0, stream>>>(Qb, Ksum, Zs);

  // KVT[n][e][d] = sum_s K[n][s][d]*V[n][s][e]  (A=K^T, B=V^T, store transposed)
  gemm_wmma<ACT_NONE, true, true, true, false>
      <<<dim3(512 / BN, 512 / BM, NB), blk, 0, stream>>>(
          Kb, 512, L * 512, Vb, 512, L * 512, KVT, 512, 512 * 512,
          (const float*)nullptr, 0, 1.0f, 1024);

  // msgt[n][l][e] = (Q[n] @ KVT[n]^T) * Z * 1024
  gemm_wmma<ACT_NONE, false, false, false, false>
      <<<dim3(512 / BN, 1024 / BM, NB), blk, 0, stream>>>(
          Qb, 512, L * 512, KVT, 512, 512 * 512, msgt, 512, L * 512,
          Zs, L, 1024.0f, 512);

  // msg2 = msgt @ Wm^T  (f32 out for layernorm)
  gemm_wmma<ACT_NONE, false, false, false, true>
      <<<dim3(512 / BN, R / BM, 1), blk, 0, stream>>>(
          msgt, 512, 0, Wmb, 512, 0, fbuf, 512, 0,
          (const float*)nullptr, 0, 1.0f, 512);

  ln_to_bf16<<<dim3((unsigned)R), blk, 0, stream>>>(fbuf, g1, b1, hcat, 1024, 512);

  // h1 = relu(hcat @ W1^T)  [65536 x 1024]
  gemm_wmma<ACT_RELU, false, false, false, false>
      <<<dim3(1024 / BN, R / BM, 1), blk, 0, stream>>>(
          hcat, 1024, 0, W1b, 1024, 0, h1, 1024, 0,
          (const float*)nullptr, 0, 1.0f, 1024);

  // h2 = h1 @ W2^T  (f32 out)
  gemm_wmma<ACT_NONE, false, false, false, true>
      <<<dim3(512 / BN, R / BM, 1), blk, 0, stream>>>(
          h1, 1024, 0, W2b, 1024, 0, fbuf, 512, 0,
          (const float*)nullptr, 0, 1.0f, 1024);

  ln_residual<<<dim3((unsigned)R), blk, 0, stream>>>(fbuf, x, g2, b2, out);
}
